// MoETransformer_6863357739242
// MI455X (gfx1250) — matrix operand order repaired and checked by reading, not compiled, not run on hardware
//
#include <hip/hip_runtime.h>
#include <hip/hip_bf16.h>

// ---------------------------------------------------------------------------
// CDNA5 (gfx1250) MoE-transformer forward.
//  - all weights converted f32->bf16 once per launch (halves HBM weight bytes)
//  - activations kept bf16 whenever the consumer is another WMMA GEMM
//  - GEMM: 128x128 block tile, 8 waves x (32x64), 8 v_wmma_f32_16x16x32_bf16
//    per wave per 32-k step, double-buffered LDS fed by
//    GLOBAL_LOAD_ASYNC_TO_LDS_B128 (ASYNCcnt) for compute/copy overlap
//  - attention: flash-style causal, Q/K tiles via async-LDS, online softmax
// ---------------------------------------------------------------------------

typedef __bf16 bf16_t;
typedef __attribute__((ext_vector_type(16))) __bf16 v16bf;
typedef __attribute__((ext_vector_type(8)))  __bf16 v8bf;
typedef __attribute__((ext_vector_type(8)))  float  v8f;

__device__ __forceinline__ bf16_t f2bf(float f) {
  union { float f; unsigned u; } c; c.f = f;
  unsigned r = c.u + 0x7FFFu + ((c.u >> 16) & 1u);   // round-to-nearest-even
  unsigned short h = (unsigned short)(r >> 16);
  return __builtin_bit_cast(bf16_t, h);
}

__device__ __forceinline__ v16bf cat8(v8bf lo, v8bf hi) {
  return __builtin_shufflevector(lo, hi, 0,1,2,3,4,5,6,7,8,9,10,11,12,13,14,15);
}

// LDS byte offset of a __shared__ pointer (generic-pointer low 32 bits).
__device__ __forceinline__ unsigned lds_off(const void* p) {
  return (unsigned)(unsigned long long)(uintptr_t)p;
}

// Async global->LDS 16B copy (per lane). GVS mode: 64-bit SGPR base + 32-bit
// unsigned VGPR byte offset. Tracked by ASYNCcnt.
__device__ __forceinline__ void async_b128(unsigned lds_byte, const void* base,
                                           unsigned gbyte) {
  asm volatile("global_load_async_to_lds_b128 %0, %1, %2"
               :
               : "v"(lds_byte), "v"(gbyte),
                 "s"((unsigned long long)(uintptr_t)base)
               : "memory");
}
__device__ __forceinline__ void wait_async0() {
  asm volatile("s_wait_asynccnt 0x0" ::: "memory");
}

// ---------------------------------------------------------------------------
// f32 -> bf16 bulk convert (grid-stride)
// ---------------------------------------------------------------------------
__global__ __launch_bounds__(256)
void cvt_kernel(const float* __restrict__ src, bf16_t* __restrict__ dst,
                size_t n) {
  size_t i = (size_t)blockIdx.x * blockDim.x + threadIdx.x;
  size_t stride = (size_t)gridDim.x * blockDim.x;
  for (; i < n; i += stride) dst[i] = f2bf(src[i]);
}

// ---------------------------------------------------------------------------
// Embedding: writes f32 residual stream and bf16 mirror for the first GEMM.
// ---------------------------------------------------------------------------
__global__ void embed_kernel(const int* __restrict__ ids,
                             const float* __restrict__ tok,
                             const float* __restrict__ pos,
                             float* __restrict__ x, bf16_t* __restrict__ xb,
                             int Sn, int Dn) {
  int i = blockIdx.x * blockDim.x + threadIdx.x;   // over B*S*D (divisible)
  int d = i % Dn;
  int t = i / Dn;
  int s = t % Sn;
  float v = tok[(size_t)ids[t] * Dn + d] + pos[(size_t)s * Dn + d];
  x[i] = v;
  xb[i] = f2bf(v);
}

// ---------------------------------------------------------------------------
// GEMM: Out[M,N] = A[M,K](bf16) @ W[N,K](bf16)^T + bias, optional ReLU.
// Output f32 (OutF) or bf16 (OutB). 128x128 tile/block, double-buffered LDS
// filled by async b128 copies; 8 waves x 8 WMMA per 32-k step.
// M,N multiples of 128; K multiple of 32.
// ---------------------------------------------------------------------------
__global__ __launch_bounds__(256)
void gemm_bf16_wmma(const bf16_t* __restrict__ A, const bf16_t* __restrict__ W,
                    const float* __restrict__ bias, float* __restrict__ OutF,
                    bf16_t* __restrict__ OutB, int M, int N, int K, int relu) {
  __shared__ __align__(16) bf16_t As[2][128][40];   // pad row to 80B (16B mult)
  __shared__ __align__(16) bf16_t Ws[2][128][40];
  (void)M;
  const int tid  = threadIdx.x;
  const int lane = tid & 31;
  const int wv   = tid >> 5;
  const int m0 = blockIdx.y * 128;
  const int n0 = blockIdx.x * 128;
  const int wm = (wv & 3) * 32;       // wave row offset (4 groups of 32)
  const int wn = (wv >> 2) * 64;      // wave col offset (2 groups of 64)
  const int hl  = lane >> 4;
  const int l15 = lane & 15;
  // loader mapping: thread -> tile row tid>>1, 32B chunk base (tid&1)*16 elems
  const int lrow = tid >> 1;
  const int lc0  = (tid & 1) * 16;

  v8f acc[8];
#pragma unroll
  for (int j = 0; j < 4; ++j) {
    float bv = bias ? bias[n0 + wn + j * 16 + l15] : 0.f;
#pragma unroll
    for (int r = 0; r < 8; ++r) { acc[j][r] = bv; acc[4 + j][r] = bv; }
  }

  auto issue = [&](int kt, int bufi) {
    unsigned la = lds_off(&As[bufi][lrow][lc0]);
    unsigned ga = (unsigned)(((size_t)(m0 + lrow) * K + kt + lc0) * 2);
    async_b128(la,      A, ga);
    async_b128(la + 16, A, ga + 16);
    unsigned lw = lds_off(&Ws[bufi][lrow][lc0]);
    unsigned gw = (unsigned)(((size_t)(n0 + lrow) * K + kt + lc0) * 2);
    async_b128(lw,      W, gw);
    async_b128(lw + 16, W, gw + 16);
  };

  issue(0, 0);
  int buf = 0;
  for (int kt = 0; kt < K; kt += 32, buf ^= 1) {
    wait_async0();        // this thread's copies for `buf` have landed
    __syncthreads();      // everyone's landed; prior reads of buf^1 retired
    if (kt + 32 < K) issue(kt + 32, buf ^ 1);   // prefetch overlaps WMMAs

    v16bf afr[2], bfr[4];
#pragma unroll
    for (int i = 0; i < 2; ++i)
      afr[i] = cat8(*(const v8bf*)&As[buf][wm + i * 16 + l15][hl * 16],
                    *(const v8bf*)&As[buf][wm + i * 16 + l15][hl * 16 + 8]);
#pragma unroll
    for (int j = 0; j < 4; ++j)
      bfr[j] = cat8(*(const v8bf*)&Ws[buf][wn + j * 16 + l15][hl * 16],
                    *(const v8bf*)&Ws[buf][wn + j * 16 + l15][hl * 16 + 8]);
#pragma unroll
    for (int i = 0; i < 2; ++i)
#pragma unroll
      for (int j = 0; j < 4; ++j)
        acc[i * 4 + j] = __builtin_amdgcn_wmma_f32_16x16x32_bf16(
            false, afr[i], false, bfr[j], (short)0, acc[i * 4 + j],
            false, false);
  }

#pragma unroll
  for (int i = 0; i < 2; ++i) {
#pragma unroll
    for (int j = 0; j < 4; ++j) {
#pragma unroll
      for (int r = 0; r < 8; ++r) {
        int row = m0 + wm + i * 16 + r + hl * 8;
        int col = n0 + wn + j * 16 + l15;
        float v = acc[i * 4 + j][r];
        if (relu) v = v > 0.f ? v : 0.f;
        if (OutB) OutB[(size_t)row * N + col] = f2bf(v);
        else      OutF[(size_t)row * N + col] = v;
      }
    }
  }
}

// ---------------------------------------------------------------------------
// Flash-style causal attention, HD=64, bf16 qkv in, bf16 out.
// One block = one (b, h, 64-query tile). Q/K tiles via async-LDS copies.
// ---------------------------------------------------------------------------
__global__ __launch_bounds__(256)
void attn_flash(const bf16_t* __restrict__ qkv, bf16_t* __restrict__ out,
                int Bn, int Sn, int Hn) {
  (void)Bn;
  const int D  = Hn * 64;
  const int qt = blockIdx.x;
  const int h  = blockIdx.y;
  const int b  = blockIdx.z;

  __shared__ __align__(16) bf16_t Qs[64][72];   // 144B rows (16B multiple)
  __shared__ __align__(16) bf16_t Ks[64][72];
  __shared__ __align__(16) bf16_t Vt[64][72];   // transposed: [d][k]
  __shared__ __align__(16) bf16_t Ps[64][72];
  __shared__ float Sls[64][65];
  __shared__ float mrow[64], lrow[64], arow[64];
  __shared__ float red[64][4];

  const int tid  = threadIdx.x;
  const int lane = tid & 31;
  const int wv   = tid >> 5;
  const int wm = (wv >> 1) * 16;
  const int wn = (wv & 1) * 32;
  const int hl  = lane >> 4;
  const int l15 = lane & 15;
  const int lr = tid >> 2;           // 0..63
  const int lc = (tid & 3) * 16;     // 16-elem chunk base

  // async copy of the Q tile (raw bf16; 1/sqrt(HD) folded into softmax)
  {
    const int s = qt * 64 + lr;
    unsigned la = lds_off(&Qs[lr][lc]);
    unsigned ga = (unsigned)(((size_t)(b * Sn + s) * 3 * D + h * 64 + lc) * 2);
    async_b128(la,      qkv, ga);
    async_b128(la + 16, qkv, ga + 16);
  }
  if (tid < 64) { mrow[tid] = -3.0e38f; lrow[tid] = 0.f; }

  v8f acc0 = {}, acc1 = {};

  for (int kt = 0; kt <= qt; ++kt) {
    __syncthreads();   // retire previous P@V reads of Vt/Ps before refill
    {
      const int s = kt * 64 + lr;
      // K tile: async copy
      unsigned la = lds_off(&Ks[lr][lc]);
      unsigned ga =
          (unsigned)(((size_t)(b * Sn + s) * 3 * D + D + h * 64 + lc) * 2);
      async_b128(la,      qkv, ga);
      async_b128(la + 16, qkv, ga + 16);
      // V tile: regular load + transpose into Vt
      const unsigned short* vp = (const unsigned short*)(qkv +
          (size_t)(b * Sn + s) * 3 * D + 2 * D + h * 64 + lc);
#pragma unroll
      for (int j = 0; j < 16; ++j)
        Vt[lc + j][lr] = __builtin_bit_cast(bf16_t, vp[j]);
    }
    wait_async0();     // K (and Q on first iteration) landed for this thread
    __syncthreads();

    // S = Q @ K^T : per wave 16x32, 2 WMMA k-steps over the 64 head dims
    v8f s0 = {}, s1 = {};
#pragma unroll
    for (int kk = 0; kk < 64; kk += 32) {
      v16bf a = cat8(*(const v8bf*)&Qs[wm + l15][kk + hl * 16],
                     *(const v8bf*)&Qs[wm + l15][kk + hl * 16 + 8]);
      v16bf b0 = cat8(*(const v8bf*)&Ks[wn + l15][kk + hl * 16],
                      *(const v8bf*)&Ks[wn + l15][kk + hl * 16 + 8]);
      v16bf b1 = cat8(*(const v8bf*)&Ks[wn + 16 + l15][kk + hl * 16],
                      *(const v8bf*)&Ks[wn + 16 + l15][kk + hl * 16 + 8]);
      s0 = __builtin_amdgcn_wmma_f32_16x16x32_bf16(false, a, false, b0,
                                                   (short)0, s0, false, false);
      s1 = __builtin_amdgcn_wmma_f32_16x16x32_bf16(false, a, false, b1,
                                                   (short)0, s1, false, false);
    }
#pragma unroll
    for (int r = 0; r < 8; ++r) {
      Sls[wm + r + hl * 8][wn + l15]      = s0[r];
      Sls[wm + r + hl * 8][wn + 16 + l15] = s1[r];
    }
    __syncthreads();

    // Online softmax (scale 0.125 = 1/sqrt(64) applied here), 4 thr/row.
    const int row = lr;
    const int q_idx = qt * 64 + row;
    float pmax = -3.0e38f;
#pragma unroll
    for (int j = 0; j < 16; ++j) {
      int c = lc + j;
      float sv = (kt * 64 + c <= q_idx) ? Sls[row][c] * 0.125f : -3.0e38f;
      pmax = fmaxf(pmax, sv);
    }
    red[row][tid & 3] = pmax;
    __syncthreads();
    if (tid < 64) {
      float mnew = mrow[tid];
#pragma unroll
      for (int t = 0; t < 4; ++t) mnew = fmaxf(mnew, red[tid][t]);
      arow[tid] = __expf(mrow[tid] - mnew);
      mrow[tid] = mnew;
    }
    __syncthreads();
    float mnew = mrow[row];
    float psum = 0.f;
#pragma unroll
    for (int j = 0; j < 16; ++j) {
      int c = lc + j;
      float p = (kt * 64 + c <= q_idx) ? __expf(Sls[row][c] * 0.125f - mnew)
                                       : 0.f;
      Ps[row][c] = f2bf(p);
      psum += p;
    }
    red[row][tid & 3] = psum;
    __syncthreads();
    if (tid < 64) {
      lrow[tid] = lrow[tid] * arow[tid] +
                  red[tid][0] + red[tid][1] + red[tid][2] + red[tid][3];
    }
    __syncthreads();

    // Rescale accumulator, then O += P @ V.
#pragma unroll
    for (int r = 0; r < 8; ++r) {
      float al = arow[wm + r + hl * 8];
      acc0[r] *= al; acc1[r] *= al;
    }
#pragma unroll
    for (int kk = 0; kk < 64; kk += 32) {
      v16bf a = cat8(*(const v8bf*)&Ps[wm + l15][kk + hl * 16],
                     *(const v8bf*)&Ps[wm + l15][kk + hl * 16 + 8]);
      v16bf b0 = cat8(*(const v8bf*)&Vt[wn + l15][kk + hl * 16],
                      *(const v8bf*)&Vt[wn + l15][kk + hl * 16 + 8]);
      v16bf b1 = cat8(*(const v8bf*)&Vt[wn + 16 + l15][kk + hl * 16],
                      *(const v8bf*)&Vt[wn + 16 + l15][kk + hl * 16 + 8]);
      acc0 = __builtin_amdgcn_wmma_f32_16x16x32_bf16(false, a, false, b0,
                                                     (short)0, acc0, false, false);
      acc1 = __builtin_amdgcn_wmma_f32_16x16x32_bf16(false, a, false, b1,
                                                     (short)0, acc1, false, false);
    }
  }

#pragma unroll
  for (int r = 0; r < 8; ++r) {
    int row = wm + r + hl * 8;
    float inv = 1.f / lrow[row];
    int s = qt * 64 + row;
    bf16_t* op = out + (size_t)(b * Sn + s) * D + h * 64;
    op[wn + l15]      = f2bf(acc0[r] * inv);
    op[wn + 16 + l15] = f2bf(acc1[r] * inv);
  }
}

// ---------------------------------------------------------------------------
// LayerNorm over D=1024: t = a + b*rowscale; LN(t)*w + bias -> f32 and/or bf16.
// ---------------------------------------------------------------------------
__global__ __launch_bounds__(256)
void ln_kernel(const float* __restrict__ a, const float* __restrict__ bsrc,
               const float* __restrict__ rowscale, const float* __restrict__ w,
               const float* __restrict__ bias, float* __restrict__ outF,
               bf16_t* __restrict__ outB, int Dn) {
  const int row = blockIdx.x;
  const int tid = threadIdx.x;
  __shared__ float red[256];
  __shared__ float mu, rstd;
  const float scl = rowscale ? rowscale[row] : 1.0f;
  const float* ap = a + (size_t)row * Dn;
  const float* bp = bsrc ? bsrc + (size_t)row * Dn : nullptr;
  float vals[4];
  float s = 0.f;
#pragma unroll
  for (int j = 0; j < 4; ++j) {
    int c = tid + j * 256;
    float v = ap[c] + (bp ? bp[c] * scl : 0.f);
    vals[j] = v; s += v;
  }
  red[tid] = s; __syncthreads();
  for (int st = 128; st > 0; st >>= 1) { if (tid < st) red[tid] += red[tid + st]; __syncthreads(); }
  if (tid == 0) mu = red[0] / (float)Dn;
  __syncthreads();
  float m = mu, vs = 0.f;
#pragma unroll
  for (int j = 0; j < 4; ++j) { float d = vals[j] - m; vs += d * d; }
  red[tid] = vs; __syncthreads();
  for (int st = 128; st > 0; st >>= 1) { if (tid < st) red[tid] += red[tid + st]; __syncthreads(); }
  if (tid == 0) rstd = rsqrtf(red[0] / (float)Dn + 1e-5f);
  __syncthreads();
  float r = rstd;
#pragma unroll
  for (int j = 0; j < 4; ++j) {
    int c = tid + j * 256;
    float o = (vals[j] - m) * r * w[c] + bias[c];
    if (outF) outF[(size_t)row * Dn + c] = o;
    if (outB) outB[(size_t)row * Dn + c] = f2bf(o);
  }
}

// ---------------------------------------------------------------------------
// Router gate (E=8): softmax, top-2 sum -> scale, argmax -> top1, probs kept.
// ---------------------------------------------------------------------------
__global__ __launch_bounds__(256)
void gate_kernel(const float* __restrict__ h, const float* __restrict__ gwp,
                 float* __restrict__ probs, float* __restrict__ scale,
                 int* __restrict__ top1, int Dn, int En) {
  const int t = blockIdx.x;
  const int tid = threadIdx.x;
  __shared__ float red[256];
  __shared__ float logit[8];
  const float* hp = h + (size_t)t * Dn;
  for (int e = 0; e < En; ++e) {
    const float* g = gwp + (size_t)e * Dn;
    float s = 0.f;
    for (int c = tid; c < Dn; c += 256) s += hp[c] * g[c];
    red[tid] = s; __syncthreads();
    for (int st = 128; st > 0; st >>= 1) { if (tid < st) red[tid] += red[tid + st]; __syncthreads(); }
    if (tid == 0) logit[e] = red[0];
    __syncthreads();
  }
  if (tid == 0) {
    float mx = logit[0];
    for (int e = 1; e < En; ++e) mx = fmaxf(mx, logit[e]);
    float p[8], se = 0.f;
    for (int e = 0; e < En; ++e) { p[e] = __expf(logit[e] - mx); se += p[e]; }
    float inv = 1.f / se;
    float v1 = -1.f, v2 = -1.f; int i1 = 0;
    for (int e = 0; e < En; ++e) {
      p[e] *= inv;
      probs[(size_t)t * En + e] = p[e];
      if (p[e] > v1) { v2 = v1; v1 = p[e]; i1 = e; }
      else if (p[e] > v2) { v2 = p[e]; }
    }
    scale[t] = v1 + v2;
    top1[t] = i1;
  }
}

__global__ __launch_bounds__(256)
void moe_reduce(const float* __restrict__ probs, const int* __restrict__ top1,
                float* __restrict__ me, float* __restrict__ ce, int T, int En) {
  const int e = blockIdx.x;
  const int tid = threadIdx.x;
  __shared__ float r1[256], r2[256];
  float s1 = 0.f, s2 = 0.f;
  for (int t = tid; t < T; t += 256) {
    s1 += probs[(size_t)t * En + e];
    if (top1[t] == e) s2 += 1.f;
  }
  r1[tid] = s1; r2[tid] = s2; __syncthreads();
  for (int st = 128; st > 0; st >>= 1) {
    if (tid < st) { r1[tid] += r1[tid + st]; r2[tid] += r2[tid + st]; }
    __syncthreads();
  }
  if (tid == 0) { me[e] = r1[0] / (float)T; ce[e] = r2[0] / (float)T; }
}

__global__ void aux_init(float* auxp) { if (threadIdx.x == 0) *auxp = 0.f; }
__global__ void aux_accum(const float* me, const float* ce, float* auxp, int En) {
  if (threadIdx.x == 0) {
    float s = 0.f;
    for (int e = 0; e < En; ++e) s += me[e] * ce[e];
    *auxp += (float)En * s;
  }
}
__global__ void aux_write(const float* auxp, float* dst) {
  if (threadIdx.x == 0) *dst = *auxp;
}

// ---------------------------------------------------------------------------
extern "C" void kernel_launch(void* const* d_in, const int* in_sizes, int n_in,
                              void* d_out, int out_size, void* d_ws, size_t ws_size,
                              hipStream_t stream) {
  (void)in_sizes; (void)n_in; (void)out_size; (void)ws_size;
  const int*   ids   = (const int*)d_in[0];
  const float* tok   = (const float*)d_in[1];
  const float* pos   = (const float*)d_in[2];
  const float* inw   = (const float*)d_in[3];
  const float* inb   = (const float*)d_in[4];
  const float* aow   = (const float*)d_in[5];
  const float* aob   = (const float*)d_in[6];
  const float* ln1w  = (const float*)d_in[7];
  const float* ln1b  = (const float*)d_in[8];
  const float* ln2w  = (const float*)d_in[9];
  const float* ln2b  = (const float*)d_in[10];
  const float* fw1   = (const float*)d_in[11];
  const float* fb1   = (const float*)d_in[12];
  const float* fw2   = (const float*)d_in[13];
  const float* fb2   = (const float*)d_in[14];
  const float* gw    = (const float*)d_in[15];
  const float* lnfw  = (const float*)d_in[16];
  const float* lnfb  = (const float*)d_in[17];
  const float* headw = (const float*)d_in[18];

  constexpr int Bc = 2, Sc = 1024, Dc = 1024, Hc = 16, DFc = 4096,
                Ec = 8, Lc = 6, Vc = 32000;
  const int T = Bc * Sc;  // 2048 tokens

  char* base = (char*)d_ws;
  size_t off = 0;
  auto alloc = [&](size_t bytes) -> void* {
    void* p = base + off;
    off = (off + bytes + 255) & ~(size_t)255;
    return p;
  };
  // bf16 weight mirrors (converted once per launch)
  bf16_t* wb_in   = (bf16_t*)alloc((size_t)Lc * 3 * Dc * Dc * 2);
  bf16_t* wb_out  = (bf16_t*)alloc((size_t)Lc * Dc * Dc * 2);
  bf16_t* wb_f1   = (bf16_t*)alloc((size_t)Lc * DFc * Dc * 2);
  bf16_t* wb_f2   = (bf16_t*)alloc((size_t)Lc * Dc * DFc * 2);
  bf16_t* wb_head = (bf16_t*)alloc((size_t)Vc * Dc * 2);
  // bf16 activations
  bf16_t* xb    = (bf16_t*)alloc((size_t)T * Dc * 2);
  bf16_t* hb    = (bf16_t*)alloc((size_t)T * Dc * 2);
  bf16_t* qkvb  = (bf16_t*)alloc((size_t)T * 3 * Dc * 2);
  bf16_t* attnb = (bf16_t*)alloc((size_t)T * Dc * 2);
  bf16_t* f1b   = (bf16_t*)alloc((size_t)T * DFc * 2);
  // f32 residual-path buffers
  float* x    = (float*)alloc((size_t)T * Dc * 4);
  float* hbuf = (float*)alloc((size_t)T * Dc * 4);
  float* buf1 = (float*)alloc((size_t)T * Dc * 4);
  float* buf2 = (float*)alloc((size_t)T * Dc * 4);
  float* probs = (float*)alloc((size_t)T * Ec * 4);
  float* scalv = (float*)alloc((size_t)T * 4);
  int*   top1  = (int*)alloc((size_t)T * 4);
  float* me   = (float*)alloc(Ec * 4);
  float* ce   = (float*)alloc(Ec * 4);
  float* auxp = (float*)alloc(4);

  // one-time (per launch) weight conversion
  cvt_kernel<<<4096, 256, 0, stream>>>(inw,   wb_in,   (size_t)Lc * 3 * Dc * Dc);
  cvt_kernel<<<4096, 256, 0, stream>>>(aow,   wb_out,  (size_t)Lc * Dc * Dc);
  cvt_kernel<<<4096, 256, 0, stream>>>(fw1,   wb_f1,   (size_t)Lc * DFc * Dc);
  cvt_kernel<<<4096, 256, 0, stream>>>(fw2,   wb_f2,   (size_t)Lc * Dc * DFc);
  cvt_kernel<<<4096, 256, 0, stream>>>(headw, wb_head, (size_t)Vc * Dc);

  aux_init<<<1, 32, 0, stream>>>(auxp);
  embed_kernel<<<(T * Dc) / 256, 256, 0, stream>>>(ids, tok, pos, x, xb, Sc, Dc);

  for (int i = 0; i < Lc; ++i) {
    // qkv (bf16) = xb @ attn_in_w^T + b
    gemm_bf16_wmma<<<dim3(3 * Dc / 128, T / 128), 256, 0, stream>>>(
        xb, wb_in + (size_t)i * 3 * Dc * Dc, inb + (size_t)i * 3 * Dc,
        nullptr, qkvb, T, 3 * Dc, Dc, 0);
    // causal attention (bf16 in/out)
    attn_flash<<<dim3(Sc / 64, Hc, Bc), 256, 0, stream>>>(qkvb, attnb, Bc, Sc, Hc);
    // out projection -> f32 (feeds residual+LN)
    gemm_bf16_wmma<<<dim3(Dc / 128, T / 128), 256, 0, stream>>>(
        attnb, wb_out + (size_t)i * Dc * Dc, aob + (size_t)i * Dc,
        buf1, nullptr, T, Dc, Dc, 0);
    // h = LN1(x + attn_out): f32 + bf16 mirror
    ln_kernel<<<T, 256, 0, stream>>>(x, buf1, nullptr,
                                     ln1w + (size_t)i * Dc, ln1b + (size_t)i * Dc,
                                     hbuf, hb, Dc);
    // ffn1 (bf16) = relu(hb @ W1^T + b1)
    gemm_bf16_wmma<<<dim3(DFc / 128, T / 128), 256, 0, stream>>>(
        hb, wb_f1 + (size_t)i * DFc * Dc, fb1 + (size_t)i * DFc,
        nullptr, f1b, T, DFc, Dc, 1);
    // ffn2 -> f32 (feeds residual+LN)
    gemm_bf16_wmma<<<dim3(Dc / 128, T / 128), 256, 0, stream>>>(
        f1b, wb_f2 + (size_t)i * Dc * DFc, fb2 + (size_t)i * Dc,
        buf2, nullptr, T, Dc, DFc, 0);

    const float* rs = nullptr;
    if (i & 1) {  // MoE gate layer
      int gi = i / 2;
      gate_kernel<<<T, 256, 0, stream>>>(hbuf, gw + (size_t)gi * Ec * Dc,
                                         probs, scalv, top1, Dc, Ec);
      moe_reduce<<<Ec, 256, 0, stream>>>(probs, top1, me, ce, T, Ec);
      aux_accum<<<1, 32, 0, stream>>>(me, ce, auxp, Ec);
      rs = scalv;
    }
    // x = LN2(h + ffn*scale): f32 + bf16 mirror for next layer
    ln_kernel<<<T, 256, 0, stream>>>(hbuf, buf2, rs,
                                     ln2w + (size_t)i * Dc, ln2b + (size_t)i * Dc,
                                     x, xb, Dc);
  }

  // final LN (bf16 mirror only) + vocab head -> logits (f32) into d_out
  ln_kernel<<<T, 256, 0, stream>>>(x, nullptr, nullptr, lnfw, lnfb,
                                   nullptr, hb, Dc);
  gemm_bf16_wmma<<<dim3(Vc / 128, T / 128), 256, 0, stream>>>(
      hb, wb_head, nullptr, (float*)d_out, nullptr, T, Vc, Dc, 0);
  aux_write<<<1, 32, 0, stream>>>(auxp, (float*)d_out + (size_t)T * Vc);
}
